// MultiHeadAttention_53446573032006
// MI455X (gfx1250) — matrix operand order repaired and checked by reading, not compiled, not run on hardware
//
#include <hip/hip_runtime.h>

// ---------------------------------------------------------------------------
// MHA forward for MI455X (gfx1250):
//   0) f32 -> bf16 conversion pass (activations + weights), bandwidth-trivial
//   1) Q/K/V projections: bf16 WMMA GEMM, tiles staged by the Tensor Data
//      Mover (tensor_load_to_lds + s_wait_tensorcnt), double-buffered LDS
//   2) Flash attention (online softmax) with v_wmma_f32_16x16x32_bf16,
//      K-tile staged by TDM, V-tile transposed on store
//   3) Output projection -> f32
// ---------------------------------------------------------------------------

typedef unsigned short u16;
typedef __attribute__((ext_vector_type(4)))  unsigned int u32x4;
typedef __attribute__((ext_vector_type(4)))  unsigned int uv4;
typedef __attribute__((ext_vector_type(4)))  int          iv4;
typedef __attribute__((ext_vector_type(8)))  int          iv8;
typedef __attribute__((ext_vector_type(16))) __bf16       v16bf;
typedef __attribute__((ext_vector_type(8)))  float        v8f;

union Frag { v16bf v; u32x4 q[2]; };

constexpr int Bc = 2, Sc = 2048, Dc = 1024, Hc = 16, DKc = 64;

#if __has_builtin(__builtin_amdgcn_tensor_load_to_lds) && \
    __has_builtin(__builtin_amdgcn_s_wait_tensorcnt)
#define HAVE_TDM 1
#if __clang_major__ >= 23
#define TDM_SIX_ARGS 1
#endif
#else
#define HAVE_TDM 0
#endif

__device__ __forceinline__ u16 f2bf(float f) {
    union { float f; unsigned u; } c; c.f = f;
    unsigned u = c.u;
    u += 0x7FFFu + ((u >> 16) & 1u);      // round-to-nearest-even
    return (u16)(u >> 16);
}

__device__ __forceinline__ v8f wmma_bf16(const Frag& a, const Frag& b, v8f c) {
    return __builtin_amdgcn_wmma_f32_16x16x32_bf16(
        /*neg_a=*/false, a.v, /*neg_b=*/false, b.v,
        /*c_mod=*/(short)0, c, /*reuse_a=*/false, /*reuse_b=*/false);
}

#if HAVE_TDM
// LDS byte offset of a __shared__ object (generic -> AS(3) -> int)
__device__ __forceinline__ unsigned lds_off(const void* p) {
    return (unsigned)(unsigned long long)
        (__attribute__((address_space(3))) const void*)p;
}

// Issue a 2-D TDM tile load: tile_d0 x tile_d1 (elements, 2B each) from a
// row-major tensor with row stride `stride_elems`, into LDS at `ldsaddr`
// with post-row padding (pad_interval/pad_amount codes per D# group1 spec).
__device__ __forceinline__ void tdm_load_2d(unsigned ldsaddr, const void* gaddr,
                                            unsigned tile_d0, unsigned tile_d1,
                                            unsigned tensor_d0, unsigned tensor_d1,
                                            unsigned stride_elems,
                                            unsigned pad_iv, unsigned pad_amt) {
    const unsigned long long ga = (unsigned long long)gaddr;
    uv4 g0;
    g0.x = 1u;                                             // count=1, user mode
    g0.y = ldsaddr;                                        // lds_addr[63:32]
    g0.z = (unsigned)(ga & 0xFFFFFFFFull);                 // global_addr lo
    g0.w = (unsigned)((ga >> 32) & 0x1FFFFFFull)           // global_addr hi
         | (2u << 30);                                     // type = 2 ("image")
    iv8 g1;
    g1[0] = (int)((1u << 16)                               // data_size = 2B
                | (1u << 20)                               // pad_enable
                | (pad_iv  << 22)                          // pad_interval
                | (pad_amt << 25));                        // pad_amount
    g1[1] = (int)((tensor_d0 & 0xFFFFu) << 16);            // tensor_dim0 lo
    g1[2] = (int)(((tensor_d0 >> 16) & 0xFFFFu)
                | ((tensor_d1 & 0xFFFFu) << 16));          // dim0 hi | dim1 lo
    g1[3] = (int)(((tensor_d1 >> 16) & 0xFFFFu)
                | ((tile_d0 & 0xFFFFu) << 16));            // dim1 hi | tile0
    g1[4] = (int)(tile_d1 & 0xFFFFu);                      // tile1 | tile2=0
    g1[5] = (int)stride_elems;                             // dim0_stride lo
    g1[6] = 0;                                             // stride hi | d1str
    g1[7] = 0;
    const iv4 z4 = {0, 0, 0, 0};
#if defined(TDM_SIX_ARGS)
    const iv8 z8 = {0, 0, 0, 0, 0, 0, 0, 0};
    __builtin_amdgcn_tensor_load_to_lds(g0, g1, z4, z4, z8, 0);
#else
    __builtin_amdgcn_tensor_load_to_lds(g0, g1, z4, z4, 0);
#endif
}
#endif  // HAVE_TDM

// ---------------------------------------------------------------------------
// f32 -> bf16 elementwise conversion (8 elems/thread, vector store)
// ---------------------------------------------------------------------------
__global__ __launch_bounds__(256)
void cvt_bf16(const float* __restrict__ in, u16* __restrict__ out, int n) {
    const int i = (blockIdx.x * 256 + threadIdx.x) * 8;
    if (i + 8 <= n) {
        union { u16 s[8]; u32x4 q; } b;
#pragma unroll
        for (int j = 0; j < 8; ++j) b.s[j] = f2bf(in[i + j]);
        *(u32x4*)(out + i) = b.q;
    }
}

// ---------------------------------------------------------------------------
// NT GEMM: C[M,N] = alpha * A[M,K] * B[N,K]^T, A/B bf16 row-major K-contig.
// Block 128x128, 256 threads (8 wave32), wave tile 64x32, K-step 64.
// LDS tile rows padded to 80 halves (160B, 16B-aligned fragment loads).
// TDM path: double-buffered, one barrier per K-step, TDM overlaps WMMA.
// ---------------------------------------------------------------------------
template <bool OUTF32>
__global__ __launch_bounds__(256)
void gemm_nt(const u16* __restrict__ A, const u16* __restrict__ Bw,
             void* __restrict__ Cout, int M, int N, int K, float alpha) {
    __shared__ u16 As[2][128 * 80];
    __shared__ u16 Bs[2][128 * 80];

    const int t    = threadIdx.x;
    const int lane = t & 31, wid = t >> 5;
    const int lr   = lane & 15, lh = lane >> 4;
    const int waveR = wid >> 2, waveC = wid & 3;
    const int bm = blockIdx.y * 128, bn = blockIdx.x * 128;

    const v8f vzero = {0.f, 0.f, 0.f, 0.f, 0.f, 0.f, 0.f, 0.f};
    v8f acc[4][2];
#pragma unroll
    for (int mi = 0; mi < 4; ++mi)
#pragma unroll
        for (int ni = 0; ni < 2; ++ni) acc[mi][ni] = vzero;

    const int iters = K / 64;

#if HAVE_TDM
    // prologue: TDM tile 0 into buffer 0
    if (wid == 0) {
        tdm_load_2d(lds_off(&As[0][0]), A + (size_t)bm * K, 64, 128,
                    (unsigned)K, (unsigned)M, (unsigned)K, /*pad_iv=*/4, /*pad_amt=*/7);
        tdm_load_2d(lds_off(&Bs[0][0]), Bw + (size_t)bn * K, 64, 128,
                    (unsigned)K, (unsigned)N, (unsigned)K, 4, 7);
    }
    int buf = 0;
    for (int it = 0; it < iters; ++it) {
        if (wid == 0) __builtin_amdgcn_s_wait_tensorcnt(0);
        __syncthreads();   // tile `it` visible; previous buffer's readers done
        if (wid == 0 && it + 1 < iters) {
            const int k1 = (it + 1) * 64;
            tdm_load_2d(lds_off(&As[buf ^ 1][0]), A + (size_t)bm * K + k1, 64, 128,
                        (unsigned)K, (unsigned)M, (unsigned)K, 4, 7);
            tdm_load_2d(lds_off(&Bs[buf ^ 1][0]), Bw + (size_t)bn * K + k1, 64, 128,
                        (unsigned)K, (unsigned)N, (unsigned)K, 4, 7);
        }
        const u16* as = &As[buf][0];
        const u16* bs = &Bs[buf][0];
#else
    const int srow = t >> 1, scb = (t & 1) * 32;  // 32 halves/thread/tile
    for (int it = 0; it < iters; ++it) {
        const int k0 = it * 64;
        {
            const u16* src = A + (size_t)(bm + srow) * K + k0 + scb;
            u32x4* dst = (u32x4*)&As[0][srow * 80 + scb];
            dst[0] = ((const u32x4*)src)[0]; dst[1] = ((const u32x4*)src)[1];
            dst[2] = ((const u32x4*)src)[2]; dst[3] = ((const u32x4*)src)[3];
        }
        {
            const u16* src = Bw + (size_t)(bn + srow) * K + k0 + scb;
            u32x4* dst = (u32x4*)&Bs[0][srow * 80 + scb];
            dst[0] = ((const u32x4*)src)[0]; dst[1] = ((const u32x4*)src)[1];
            dst[2] = ((const u32x4*)src)[2]; dst[3] = ((const u32x4*)src)[3];
        }
        if (it + 1 < iters) {
            __builtin_prefetch(A  + (size_t)(bm + srow) * K + k0 + 64 + scb, 0, 1);
            __builtin_prefetch(Bw + (size_t)(bn + srow) * K + k0 + 64 + scb, 0, 1);
        }
        __syncthreads();
        const u16* as = &As[0][0];
        const u16* bs = &Bs[0][0];
#endif
        // ---- compute: 16 WMMAs per K-step (2 sub-steps of 32) ----
        Frag bfr[2][2];
#pragma unroll
        for (int ni = 0; ni < 2; ++ni)
#pragma unroll
            for (int ks = 0; ks < 2; ++ks) {
                const u16* p = &bs[(waveC * 32 + ni * 16 + lr) * 80 + ks * 32 + lh * 16];
                bfr[ni][ks].q[0] = *(const u32x4*)p;
                bfr[ni][ks].q[1] = *(const u32x4*)(p + 8);
            }
#pragma unroll
        for (int mi = 0; mi < 4; ++mi) {
            const u16* p = &as[(waveR * 64 + mi * 16 + lr) * 80];
            Frag a0, a1;
            a0.q[0] = *(const u32x4*)(p + lh * 16);
            a0.q[1] = *(const u32x4*)(p + lh * 16 + 8);
            a1.q[0] = *(const u32x4*)(p + 32 + lh * 16);
            a1.q[1] = *(const u32x4*)(p + 32 + lh * 16 + 8);
            acc[mi][0] = wmma_bf16(a0, bfr[0][0], acc[mi][0]);
            acc[mi][0] = wmma_bf16(a1, bfr[0][1], acc[mi][0]);
            acc[mi][1] = wmma_bf16(a0, bfr[1][0], acc[mi][1]);
            acc[mi][1] = wmma_bf16(a1, bfr[1][1], acc[mi][1]);
        }
#if HAVE_TDM
        buf ^= 1;
    }
#else
        __syncthreads();
    }
#endif

    // epilogue: C layout -> col = lane&15, row = r + 8*(lane>>4)
#pragma unroll
    for (int mi = 0; mi < 4; ++mi)
#pragma unroll
        for (int ni = 0; ni < 2; ++ni) {
            const int gn = bn + waveC * 32 + ni * 16 + lr;
#pragma unroll
            for (int r = 0; r < 8; ++r) {
                const int gm = bm + waveR * 64 + mi * 16 + r + 8 * lh;
                const float val = acc[mi][ni][r] * alpha;
                if constexpr (OUTF32)
                    ((float*)Cout)[(size_t)gm * N + gn] = val;
                else
                    ((u16*)Cout)[(size_t)gm * N + gn] = f2bf(val);
            }
        }
}

// ---------------------------------------------------------------------------
// Flash attention: block = 4 waves = 64 query rows of one (b,h).
// Q pre-scaled by 1/sqrt(DK). Causal mask applied analytically (mask == tril).
// ---------------------------------------------------------------------------
__global__ __launch_bounds__(128)
void attn_fwd(const u16* __restrict__ Q, const u16* __restrict__ Km,
              const u16* __restrict__ V, u16* __restrict__ O) {
    __shared__ u16 Ks[64 * 80];       // K tile:  [kv][dk]  (TDM-staged)
    __shared__ u16 Vs[64 * 80];       // V tile transposed: [dk][kv]
    __shared__ u16 Ps[4 * 16 * 80];   // per-wave P staging (C-layout -> A-layout)

    const int t    = threadIdx.x;
    const int lane = t & 31, wid = t >> 5;
    const int lr   = lane & 15, lh = lane >> 4;
    const int qt = blockIdx.x, h = blockIdx.y, b = blockIdx.z;
    const int q0 = qt * 64;
    const size_t rowbase = (size_t)b * Sc;
    const int colbase = h * DKc;

    // Q fragments for this wave's 16 rows (register-resident whole kernel)
    Frag qf[2];
    {
        const u16* qp = Q + (rowbase + q0 + wid * 16 + lr) * Dc + colbase;
#pragma unroll
        for (int ks = 0; ks < 2; ++ks) {
            qf[ks].q[0] = *(const u32x4*)(qp + ks * 32 + lh * 16);
            qf[ks].q[1] = *(const u32x4*)(qp + ks * 32 + lh * 16 + 8);
        }
    }

    const v8f vzero = {0.f, 0.f, 0.f, 0.f, 0.f, 0.f, 0.f, 0.f};
    v8f oacc[4];
#pragma unroll
    for (int jo = 0; jo < 4; ++jo) oacc[jo] = vzero;
    float mst[8], lst[8];
#pragma unroll
    for (int r = 0; r < 8; ++r) { mst[r] = -1e30f; lst[r] = 0.f; }

    const int srow = t >> 1, scb = (t & 1) * 32;
    const int ntiles = qt + 1;                     // causal: skip future tiles

    for (int kt = 0; kt < ntiles; ++kt) {
        const int kv0 = kt * 64;
#if HAVE_TDM
        if (wid == 0)   // K tile via Tensor Data Mover (row stride Dc)
            tdm_load_2d(lds_off(&Ks[0]),
                        Km + (rowbase + kv0) * Dc + colbase, 64, 64,
                        (unsigned)Dc, (unsigned)(Bc * Sc), (unsigned)Dc, 4, 7);
#else
        {
            const u16* src = Km + (rowbase + kv0 + srow) * Dc + colbase + scb;
            u32x4* dst = (u32x4*)&Ks[srow * 80 + scb];
            dst[0] = ((const u32x4*)src)[0]; dst[1] = ((const u32x4*)src)[1];
            dst[2] = ((const u32x4*)src)[2]; dst[3] = ((const u32x4*)src)[3];
        }
#endif
        {   // V tile transposed on store ([dk][kv]) for B-frag K-contiguity
            const u16* src = V + (rowbase + kv0 + srow) * Dc + colbase + scb;
            union { u16 s[32]; u32x4 q[4]; } buf;
            buf.q[0] = ((const u32x4*)src)[0];
            buf.q[1] = ((const u32x4*)src)[1];
            buf.q[2] = ((const u32x4*)src)[2];
            buf.q[3] = ((const u32x4*)src)[3];
#pragma unroll
            for (int i = 0; i < 32; ++i) Vs[(scb + i) * 80 + srow] = buf.s[i];
        }
        if (kt + 1 < ntiles)
            __builtin_prefetch(V + (rowbase + kv0 + 64 + srow) * Dc + colbase + scb, 0, 1);
#if HAVE_TDM
        if (wid == 0) __builtin_amdgcn_s_wait_tensorcnt(0);
#endif
        __syncthreads();

        // scores S = Q K^T (16 q-rows x 64 kv-cols), DK = 2 wmma k-steps
        v8f s[4];
#pragma unroll
        for (int j = 0; j < 4; ++j) {
            Frag kb0, kb1;
            const u16* p = &Ks[(j * 16 + lr) * 80];
            kb0.q[0] = *(const u32x4*)(p + lh * 16);
            kb0.q[1] = *(const u32x4*)(p + lh * 16 + 8);
            kb1.q[0] = *(const u32x4*)(p + 32 + lh * 16);
            kb1.q[1] = *(const u32x4*)(p + 32 + lh * 16 + 8);
            v8f z = vzero;
            z = wmma_bf16(qf[0], kb0, z);
            z = wmma_bf16(qf[1], kb1, z);
            s[j] = z;
        }

        if (kt == qt) {  // diagonal tile: causal mask
#pragma unroll
            for (int j = 0; j < 4; ++j)
#pragma unroll
                for (int r = 0; r < 8; ++r) {
                    const int qg = q0 + wid * 16 + r + 8 * lh;
                    const int kg = kv0 + j * 16 + lr;
                    s[j][r] = (kg > qg) ? -1e9f : s[j][r];
                }
        }

        // online softmax (rows live in 16-lane column groups -> shfl_xor)
#pragma unroll
        for (int r = 0; r < 8; ++r) {
            float tm = fmaxf(fmaxf(s[0][r], s[1][r]), fmaxf(s[2][r], s[3][r]));
            tm = fmaxf(tm, __shfl_xor(tm, 1, 32));
            tm = fmaxf(tm, __shfl_xor(tm, 2, 32));
            tm = fmaxf(tm, __shfl_xor(tm, 4, 32));
            tm = fmaxf(tm, __shfl_xor(tm, 8, 32));
            const float mnew = fmaxf(mst[r], tm);
            const float sc = __expf(mst[r] - mnew);
            mst[r] = mnew;
            lst[r] *= sc;
#pragma unroll
            for (int jo = 0; jo < 4; ++jo) oacc[jo][r] *= sc;
            float ts = 0.f;
#pragma unroll
            for (int j = 0; j < 4; ++j) {
                const float p = __expf(s[j][r] - mnew);
                s[j][r] = p;
                ts += p;
            }
            ts += __shfl_xor(ts, 1, 32);
            ts += __shfl_xor(ts, 2, 32);
            ts += __shfl_xor(ts, 4, 32);
            ts += __shfl_xor(ts, 8, 32);
            lst[r] += ts;
        }

        // P (C-layout) -> wave-private LDS -> A-layout (LDS in-order per wave)
        u16* pw = &Ps[wid * 16 * 80];
#pragma unroll
        for (int j = 0; j < 4; ++j)
#pragma unroll
            for (int r = 0; r < 8; ++r)
                pw[(r + 8 * lh) * 80 + j * 16 + lr] = f2bf(s[j][r]);

        // O += P @ V   (K dim = 64 kv -> 2 wmma steps, 4 dk fragments)
#pragma unroll
        for (int ks = 0; ks < 2; ++ks) {
            Frag pf;
            const u16* pp = &Ps[wid * 16 * 80 + lr * 80 + ks * 32];
            pf.q[0] = *(const u32x4*)(pp + lh * 16);
            pf.q[1] = *(const u32x4*)(pp + lh * 16 + 8);
#pragma unroll
            for (int jo = 0; jo < 4; ++jo) {
                Frag vb;
                const u16* vp = &Vs[(jo * 16 + lr) * 80 + ks * 32 + lh * 16];
                vb.q[0] = *(const u32x4*)vp;
                vb.q[1] = *(const u32x4*)(vp + 8);
                oacc[jo] = wmma_bf16(pf, vb, oacc[jo]);
            }
        }
        __syncthreads();
    }

    // epilogue: O = acc / l, bf16 to workspace
#pragma unroll
    for (int jo = 0; jo < 4; ++jo)
#pragma unroll
        for (int r = 0; r < 8; ++r) {
            const int qg = q0 + wid * 16 + r + 8 * lh;
            const float val = oacc[jo][r] / lst[r];
            O[(rowbase + qg) * Dc + colbase + jo * 16 + lr] = f2bf(val);
        }
}

// ---------------------------------------------------------------------------
extern "C" void kernel_launch(void* const* d_in, const int* in_sizes, int n_in,
                              void* d_out, int out_size, void* d_ws, size_t ws_size,
                              hipStream_t stream) {
    (void)in_sizes; (void)n_in; (void)out_size; (void)ws_size;

    const float* q  = (const float*)d_in[0];
    const float* k  = (const float*)d_in[1];
    const float* v  = (const float*)d_in[2];
    // d_in[3]: mask (1,S,S) int32 == tril -> causality computed analytically.
    const float* Wq = (const float*)d_in[4];
    const float* Wk = (const float*)d_in[5];
    const float* Wv = (const float*)d_in[6];
    const float* Wo = (const float*)d_in[7];

    const size_t eAct = (size_t)Bc * Sc * Dc;    // 4,194,304
    const size_t eW   = (size_t)Dc * Dc;         // 1,048,576
    u16* Xq = (u16*)d_ws;                        // bf16 workspace (64 MiB)
    u16* Xk = Xq + eAct;
    u16* Xv = Xk + eAct;
    u16* Wqb = Xv + eAct;
    u16* Wkb = Wqb + eW;
    u16* Wvb = Wkb + eW;
    u16* Wob = Wvb + eW;
    u16* Qw  = Wob + eW;
    u16* Kw  = Qw + eAct;
    u16* Vw  = Kw + eAct;
    u16* Ow  = Vw + eAct;

    const int gAct = (int)(eAct / (256 * 8));    // 2048 blocks
    const int gW   = (int)(eW   / (256 * 8));    // 512 blocks
    cvt_bf16<<<gAct, 256, 0, stream>>>(q,  Xq,  (int)eAct);
    cvt_bf16<<<gAct, 256, 0, stream>>>(k,  Xk,  (int)eAct);
    cvt_bf16<<<gAct, 256, 0, stream>>>(v,  Xv,  (int)eAct);
    cvt_bf16<<<gW,   256, 0, stream>>>(Wq, Wqb, (int)eW);
    cvt_bf16<<<gW,   256, 0, stream>>>(Wk, Wkb, (int)eW);
    cvt_bf16<<<gW,   256, 0, stream>>>(Wv, Wvb, (int)eW);
    cvt_bf16<<<gW,   256, 0, stream>>>(Wo, Wob, (int)eW);

    const dim3 ggrid(Dc / 128, (Bc * Sc) / 128); // (8, 32)
    const float inv_sqrt_dk = 0.125f;            // folded into Q projection

    gemm_nt<false><<<ggrid, 256, 0, stream>>>(Xq, Wqb, Qw, Bc * Sc, Dc, Dc, inv_sqrt_dk);
    gemm_nt<false><<<ggrid, 256, 0, stream>>>(Xk, Wkb, Kw, Bc * Sc, Dc, Dc, 1.0f);
    gemm_nt<false><<<ggrid, 256, 0, stream>>>(Xv, Wvb, Vw, Bc * Sc, Dc, Dc, 1.0f);

    attn_fwd<<<dim3(Sc / 64, Hc, Bc), 128, 0, stream>>>(Qw, Kw, Vw, Ow);

    gemm_nt<true><<<ggrid, 256, 0, stream>>>(Ow, Wob, d_out, Bc * Sc, Dc, Dc, 1.0f);
}
